// SFA_46188078301688
// MI455X (gfx1250) — compile-verified, matrix operand order checked by reading, not run on hardware
//
#include <hip/hip_runtime.h>

#define VNUM 2
#define CNUM 64
#define HNUM 128
#define WNUM 416
#define HWNUM (HNUM * WNUM)       // 53248
#define NNUM 262144
#define PNUM 4
#define EPS2 1e-16f               // EPS*EPS, EPS = 1e-8

typedef __attribute__((ext_vector_type(16))) _Float16 v16h;
typedef __attribute__((ext_vector_type(8)))  _Float16 v8h;
typedef __attribute__((ext_vector_type(8)))  float    v8f;

// ---------------------------------------------------------------------------
// Kernel 1: transpose x2d (V, C, HW) -> xT (V, HW, C) so each pixel's 64
// channels become one contiguous 256B row (all gathers then L2-resident:
// whole tensor is 27MB << 192MB L2).
// ---------------------------------------------------------------------------
__global__ __launch_bounds__(256) void sfa_transpose_kernel(
    const float* __restrict__ x2d, float* __restrict__ xT) {
  __shared__ float tile[64][65];
  const int blk  = blockIdx.x;            // 0 .. 2*832-1
  const int v    = blk / (HWNUM / 64);
  const int p0   = (blk % (HWNUM / 64)) * 64;
  const int tid  = threadIdx.x;           // 256 threads
  const float* src = x2d + (size_t)v * CNUM * HWNUM;
  float*       dst = xT  + (size_t)v * HWNUM * CNUM;
#pragma unroll
  for (int i = 0; i < 16; ++i) {
    int c = i * 4 + (tid >> 6);
    int p = tid & 63;
    tile[c][p] = src[(size_t)c * HWNUM + p0 + p];   // coalesced 256B reads
  }
  __syncthreads();
#pragma unroll
  for (int i = 0; i < 16; ++i) {
    int p = i * 4 + (tid >> 6);
    int c = tid & 63;
    dst[(size_t)(p0 + p) * CNUM + c] = tile[c][p];  // coalesced 256B writes
  }
}

// ---------------------------------------------------------------------------
// WMMA helpers: row-sums of a 16x64 elementwise-product tile via
// D = (A.*B) x ones(32x16) + C, accumulated f32 across two K=32 chunks.
// A ones-valued B-operand is layout independent -> no B shuffling needed.
// 16-bit A layout: lanes 0-15 hold K=cb..cb+7 (halves 0-7) and K=cb+16..cb+23
// (halves 8-15); lanes 16-31 the +8 K-offset variant.
// ---------------------------------------------------------------------------
__device__ __forceinline__ v16h sfa_ldfrag(const _Float16 (*Fv)[72], int m,
                                           int cb) {
  v8h lo = *(const v8h*)&Fv[m][cb];
  v8h hi = *(const v8h*)&Fv[m][cb + 16];
  return __builtin_shufflevector(lo, hi, 0, 1, 2, 3, 4, 5, 6, 7, 8, 9, 10, 11,
                                 12, 13, 14, 15);
}

__device__ __forceinline__ v8f sfa_rowsum_prod(const _Float16 (*A)[72],
                                               const _Float16 (*B)[72],
                                               int lane) {
  v16h ones;
#pragma unroll
  for (int i = 0; i < 16; ++i) ones[i] = (_Float16)1.0f;
  const int m  = lane & 15;
  const int hs = (lane >> 4) << 3;
  v8f acc = {};
#pragma unroll
  for (int q = 0; q < 2; ++q) {                     // channels [0,32), [32,64)
    const int cb = q * 32 + hs;
    v16h a = sfa_ldfrag(A, m, cb);
    v16h b = sfa_ldfrag(B, m, cb);                  // CSE'd when A==B
    v16h p = a * b;                                 // packed f16 products
    acc = __builtin_amdgcn_wmma_f32_16x16x32_f16(false, p, false, ones,
                                                 (short)0, acc, false, false);
  }
  return acc;
}

// ---------------------------------------------------------------------------
// Kernel 2: main SFA. Block = 512 threads = 16 waves; wave w owns voxel
// n = blockIdx*16 + w. Lane owns channels (2*lane, 2*lane+1).
// Phase A: vectorized idx/mask loads, masked gather + average -> feats
//          (f32 regs), staged as f16 in LDS.
// Phase B: waves 0..2 compute dot / |f0|^2 / |f1|^2 row-sums in parallel
//          with v_wmma_f32_16x16x32_f16.
// Phase C: per-wave combine; outputs staged in LDS, then 256 coalesced
//          b128 stores along N.
// ---------------------------------------------------------------------------
__global__ __launch_bounds__(512) void sfa_main_kernel(
    const float* __restrict__ x2d,                  // (V, C, HW)
    const float* __restrict__ xT,                   // (V, HW, C) or unused
    const int* __restrict__ pp,                     // (V, N, P, 2) int32
    const unsigned char* __restrict__ mask,         // (V, N, P) bool
    float* __restrict__ out,                        // (C, N)
    int useT) {
  __shared__ __align__(16) _Float16 F[2][16][72];   // [view][nLocal][chan]
  __shared__ float sums[3][16];                     // dot, n0, n1 per voxel
  __shared__ __align__(16) float tileO[64][20];     // [chan][nLocal] padded

  const int tid  = threadIdx.x;
  const int lane = tid & 31;
  const int wid  = tid >> 5;                        // 0..15
  const int n0b  = blockIdx.x * 16;
  const int n    = n0b + wid;
  const int c0   = lane * 2;

  float feat[2][2];
  float wv[2];

#pragma unroll
  for (int v = 0; v < VNUM; ++v) {
    const size_t ib = ((size_t)v * NNUM + n) * PNUM;
    // 4 mask bytes as one u32; 8 coords as two int4 (32B, aligned).
    const unsigned int mw = *(const unsigned int*)(mask + ib);
    const int4 q0 = ((const int4*)(pp + ib * 2))[0];  // p0:(x,y) p1:(x,y)
    const int4 q1 = ((const int4*)(pp + ib * 2))[1];  // p2:(x,y) p3:(x,y)
    const int pix[PNUM] = {q0.y * WNUM + q0.x, q0.w * WNUM + q0.z,
                           q1.y * WNUM + q1.x, q1.w * WNUM + q1.z};
    float a0 = 0.f, a1 = 0.f;
    int cnt = 0;
#pragma unroll
    for (int p = 0; p < PNUM; ++p) {
      if ((mw >> (8 * p)) & 0xFFu) {
        float t0, t1;
        if (useT) {
          const float2 t =
              *(const float2*)(xT + ((size_t)v * HWNUM + pix[p]) * CNUM + c0);
          t0 = t.x; t1 = t.y;
        } else {
          const float* s =
              x2d + ((size_t)v * CNUM + c0) * (size_t)HWNUM + pix[p];
          t0 = s[0];
          t1 = s[HWNUM];
        }
        a0 += t0; a1 += t1; ++cnt;
      }
    }
    const float w   = cnt > 0 ? 1.f : 0.f;
    const float inv = cnt > 0 ? 1.f / (float)cnt : 0.f;
    feat[v][0] = a0 * inv;
    feat[v][1] = a1 * inv;
    wv[v] = w;
    F[v][wid][c0]     = (_Float16)feat[v][0];
    F[v][wid][c0 + 1] = (_Float16)feat[v][1];
  }
  __syncthreads();

  // --- Phase B: three independent WMMA reductions on waves 0,1,2 ----------
  if (wid < 3) {
    v8f acc;
    if (wid == 0)      acc = sfa_rowsum_prod(F[0], F[1], lane);  // dot
    else if (wid == 1) acc = sfa_rowsum_prod(F[0], F[0], lane);  // |f0|^2
    else               acc = sfa_rowsum_prod(F[1], F[1], lane);  // |f1|^2
    // D layout: lane 0 VGPR r = row M=r (N=0); lane 16 VGPR r = row M=r+8.
    if (lane == 0) {
#pragma unroll
      for (int r = 0; r < 8; ++r) sums[wid][r] = acc[r];
    }
    if (lane == 16) {
#pragma unroll
      for (int r = 0; r < 8; ++r) sums[wid][8 + r] = acc[r];
    }
  }
  __syncthreads();

  // --- Phase C: per-voxel combine -----------------------------------------
  const float dot  = sums[0][wid];
  const float nsq  = sums[1][wid] * sums[2][wid];
  const float cosv = dot * rsqrtf(fmaxf(nsq, EPS2));
  const float wij  = wv[0] * wv[1];
  const float diff = wv[0] - wv[1];
  const float wi   = diff > 0.f ? 1.f : 0.f;
  const float wj   = diff < 0.f ? 1.f : 0.f;
  const float cw   = cosv * wij;
  const float g0   = 0.5f * (cw + wi);              // / (V*(V-1)) = /2
  const float g1   = 0.5f * (cw + wj);

  tileO[c0][wid]     = g0 * feat[0][0] + g1 * feat[1][0];
  tileO[c0 + 1][wid] = g0 * feat[0][1] + g1 * feat[1][1];
  __syncthreads();

  // --- Coalesced b128 stores: 256 threads x float4 along N ---------------
  if (tid < 256) {
    const int c = tid >> 2;
    const int j = (tid & 3) * 4;
    const float4 val = *(const float4*)&tileO[c][j];
    *(float4*)(out + (size_t)c * NNUM + n0b + j) = val;
  }
}

// ---------------------------------------------------------------------------
extern "C" void kernel_launch(void* const* d_in, const int* in_sizes, int n_in,
                              void* d_out, int out_size, void* d_ws,
                              size_t ws_size, hipStream_t stream) {
  const float*         x2d  = (const float*)d_in[0];
  const int*           pp   = (const int*)d_in[1];
  const unsigned char* mask = (const unsigned char*)d_in[2];
  float*               out  = (float*)d_out;
  float*               xT   = (float*)d_ws;

  const size_t need = (size_t)VNUM * HWNUM * CNUM * sizeof(float);  // ~27.3 MB
  const int useT = (ws_size >= need) ? 1 : 0;

  if (useT) {
    sfa_transpose_kernel<<<VNUM * (HWNUM / 64), 256, 0, stream>>>(x2d, xT);
  }
  sfa_main_kernel<<<NNUM / 16, 512, 0, stream>>>(x2d, xT, pp, mask, out, useT);
}
